// StackoverflowNet_53919019434256
// MI455X (gfx1250) — matrix operand order repaired and checked by reading, not compiled, not run on hardware
//
#include <hip/hip_runtime.h>

#define Vv 10000
#define Ee 256
#define Hh 512
#define Ss 25
#define Bb 128
#define H4 2048
#define SV 250000  // Ss*Vv

typedef __attribute__((ext_vector_type(16))) __bf16 v16bf;
typedef __attribute__((ext_vector_type(8)))  __bf16 v8bf;
typedef __attribute__((ext_vector_type(8)))  float  v8f;

__device__ __forceinline__ float sigf(float x){ return 1.0f / (1.0f + __expf(-x)); }

// ------------------------------------------------------------------ init
__global__ void init_zero_kernel(float* gc, float* lc, __bf16* ghbf, __bf16* lhbf, float* acc){
  int i = blockIdx.x * blockDim.x + threadIdx.x;
  if (i < Bb * Hh){ gc[i] = 0.f; lc[i] = 0.f; ghbf[i] = (__bf16)0.f; lhbf[i] = (__bf16)0.f; }
  if (i < 2 * Bb) acc[i] = 0.f;
}

// ------------------------------------------------------------------ embed
__global__ void embed_kernel(const int* __restrict__ x, const float* __restrict__ emb,
                             float* __restrict__ e32, __bf16* __restrict__ ebf){
  int row = blockIdx.x;            // b*S+s
  int k   = threadIdx.x;           // 0..E-1
  int tok = x[row];
  float v = emb[(size_t)tok * Ee + k];
  e32[(size_t)row * Ee + k] = v;
  ebf[(size_t)row * Ee + k] = (__bf16)v;
}

// ------------------------------------------------------------------ f32 -> bf16
__global__ void cvt_bf16_kernel(const float* __restrict__ src, __bf16* __restrict__ dst, int n){
  int i = blockIdx.x * blockDim.x + threadIdx.x;
  if (i < n) dst[i] = (__bf16)src[i];
}

// ------------------------------------------------------------------ WMMA GEMM
// C[M,N] = A[M,K](bf16,row-major) x W[N,K](bf16,row-major, used as B columns)
//          [+ bias[N]] [+ add[row*addLd + n]]
// grid.z selects the {A,W,bias,add,C} set (two weight banks per launch).
// One wave computes a 16 x (NT*16) strip; the grid covers the tile space
// EXACTLY (no early-out, no edge guards) so the body is branch-free:
// no accumulator phi-copies, EXEC all-ones throughout.
template <int K, int NT, bool HAS_BIAS, bool HAS_ADD>
__global__ __launch_bounds__(256)
void gemm_bf16_wmma(const __bf16* __restrict__ A0, const __bf16* __restrict__ A1, int lda,
                    const __bf16* __restrict__ W0, const __bf16* __restrict__ W1, int ldw,
                    const float* __restrict__ bias0, const float* __restrict__ bias1,
                    const float* __restrict__ add0, const float* __restrict__ add1, int addLd,
                    float* __restrict__ C0, float* __restrict__ C1, int ldc,
                    int mtiles, int nbase){
  const int lane = threadIdx.x & 31;
  // wave-uniform tile id, forced scalar
  const int wid = __builtin_amdgcn_readfirstlane(blockIdx.x * 8 + (threadIdx.x >> 5));
  const int mT = wid % mtiles;
  const int nG = wid / mtiles;
  const int z  = blockIdx.z;
  const __bf16* A   = (z && A1) ? A1 : A0;
  const __bf16* W   = z ? W1 : W0;
  const float* bias = z ? bias1 : bias0;
  const float* add  = z ? add1  : add0;
  float* C          = z ? C1 : C0;
  const int m0 = mT << 4;
  const int n0 = nbase + nG * (NT * 16);
  const int l   = lane & 15;
  const int hiA = (lane >> 4) << 3;   // A: lanes16-31 hold K+8..15 / K+24..31
  const int hiB = (lane >> 4) << 3;   // B: lanes16-31 hold K+8..15 / K+24..31

  v8f acc[NT] = {};
  const __bf16* arow  = A + (size_t)(m0 + l) * lda + hiA;
  const __bf16* wbase = W + (size_t)(n0 + l) * ldw + hiB;
#pragma unroll
  for (int k0 = 0; k0 < K; k0 += 32){
    v8bf alo = *reinterpret_cast<const v8bf*>(arow + k0);
    v8bf ahi = *reinterpret_cast<const v8bf*>(arow + k0 + 16);
    v16bf af = __builtin_shufflevector(alo, ahi, 0,1,2,3,4,5,6,7,8,9,10,11,12,13,14,15);
    v16bf bfr[NT];
    // issue all B-fragment loads for this K-chunk first (batches into a clause)
#pragma unroll
    for (int j = 0; j < NT; ++j){
      const __bf16* brow = wbase + (size_t)(j << 4) * ldw + k0;
      v8bf blo = *reinterpret_cast<const v8bf*>(brow);
      v8bf bhi = *reinterpret_cast<const v8bf*>(brow + 16);
      bfr[j] = __builtin_shufflevector(blo, bhi, 0,1,2,3,4,5,6,7,8,9,10,11,12,13,14,15);
    }
#pragma unroll
    for (int j = 0; j < NT; ++j){
      acc[j] = __builtin_amdgcn_wmma_f32_16x16x32_bf16(
          false, af, false, bfr[j], (short)0, acc[j], false, false);
    }
  }
  const int rbase = m0 + ((lane >> 4) << 3);      // C: VGPR r -> row rbase+r, col n0+j*16+l
#pragma unroll
  for (int j = 0; j < NT; ++j){
    const int n = n0 + (j << 4) + l;
    const float bv = HAS_BIAS ? bias[n] : 0.f;
#pragma unroll
    for (int r = 0; r < 8; ++r){
      const int row = rbase + r;
      float v = acc[j][r] + bv;
      if (HAS_ADD) v += add[(size_t)row * addLd + n];
      C[(size_t)row * ldc + n] = v;
    }
  }
}

// ------------------------------------------------------------------ LSTM cell update (per step)
// gates_* = e@Wih.T + h@Whh.T + b (already formed). Computes both cells,
// the pointer softmax over feat=[e, gh2, lh2], mixes states, emits bf16 h.
__global__ __launch_bounds__(256)
void lstm_cell_update(const float* __restrict__ gates_g, const float* __restrict__ gates_l,
                      const float* __restrict__ e32,
                      const float* __restrict__ plW, const float* __restrict__ plb,
                      float* __restrict__ gc, float* __restrict__ lc,
                      __bf16* __restrict__ ghbf, __bf16* __restrict__ lhbf,
                      __bf16* __restrict__ hsbf, int t){
  __shared__ float sh_hg[Hh], sh_hl[Hh], sh_cg[Hh], sh_cl[Hh];
  __shared__ float red0[8], red1[8], spr[2];
  const int b = blockIdx.x, tid = threadIdx.x;
  const float* Gg = gates_g + (size_t)b * H4;
  const float* Gl = gates_l + (size_t)b * H4;
  for (int j = tid; j < Hh; j += 256){
    { float i_ = Gg[j], f_ = Gg[Hh + j], g_ = Gg[2*Hh + j], o_ = Gg[3*Hh + j];
      float c2 = sigf(f_) * gc[b*Hh + j] + sigf(i_) * tanhf(g_);
      sh_cg[j] = c2; sh_hg[j] = sigf(o_) * tanhf(c2); }
    { float i_ = Gl[j], f_ = Gl[Hh + j], g_ = Gl[2*Hh + j], o_ = Gl[3*Hh + j];
      float c2 = sigf(f_) * lc[b*Hh + j] + sigf(i_) * tanhf(g_);
      sh_cl[j] = c2; sh_hl[j] = sigf(o_) * tanhf(c2); }
  }
  __syncthreads();
  // feat . plW rows (feat dim = E + 2H = 1280)
  float d0 = 0.f, d1 = 0.f;
  const float* e = e32 + ((size_t)b * Ss + t) * Ee;
  for (int i = tid; i < Ee + 2*Hh; i += 256){
    float v = (i < Ee) ? e[i] : (i < Ee + Hh ? sh_hg[i - Ee] : sh_hl[i - Ee - Hh]);
    d0 += v * plW[i]; d1 += v * plW[Ee + 2*Hh + i];
  }
  for (int o = 16; o > 0; o >>= 1){ d0 += __shfl_xor(d0, o, 32); d1 += __shfl_xor(d1, o, 32); }
  if ((tid & 31) == 0){ red0[tid >> 5] = d0; red1[tid >> 5] = d1; }
  __syncthreads();
  if (tid == 0){
    float a0 = plb[0], a1 = plb[1];
    for (int w = 0; w < 8; ++w){ a0 += red0[w]; a1 += red1[w]; }
    float m = fmaxf(a0, a1), e0 = __expf(a0 - m), e1 = __expf(a1 - m), inv = 1.f/(e0 + e1);
    spr[0] = e0 * inv; spr[1] = e1 * inv;
  }
  __syncthreads();
  const float p0 = spr[0], p1 = spr[1];     // p0 pairs with LOCAL, p1 with GLOBAL
  for (int j = tid; j < Hh; j += 256){
    float nh = p0 * sh_hl[j] + p1 * sh_hg[j];
    float nc = p0 * sh_cl[j] + p1 * sh_cg[j];
    gc[b*Hh + j] = nc;            lc[b*Hh + j] = sh_cl[j];
    ghbf[b*Hh + j] = (__bf16)nh;  lhbf[b*Hh + j] = (__bf16)sh_hl[j];
    hsbf[((size_t)b * Ss + t) * Hh + j] = (__bf16)nh;
  }
}

// ------------------------------------------------------------------ pr1
__global__ __launch_bounds__(256)
void pr1_kernel(const float* __restrict__ gx, const float* __restrict__ lx,
                const float* __restrict__ p1W, const float* __restrict__ p1b,
                float* __restrict__ pr1){
  const int b = blockIdx.x, tid = threadIdx.x;
  __shared__ float red0[8], red1[8];
  const int SE = Ss * Ee;                  // 6400
  const float* g = gx + (size_t)b * SE;
  const float* l = lx + (size_t)b * SE;
  float d0 = 0.f, d1 = 0.f;
  for (int i = tid; i < SE; i += 256){
    float gv = g[i], lv = l[i];
    d0 += gv * p1W[i]          + lv * p1W[SE + i];
    d1 += gv * p1W[2*SE + i]   + lv * p1W[3*SE + i];
  }
  for (int o = 16; o > 0; o >>= 1){ d0 += __shfl_xor(d0, o, 32); d1 += __shfl_xor(d1, o, 32); }
  if ((tid & 31) == 0){ red0[tid >> 5] = d0; red1[tid >> 5] = d1; }
  __syncthreads();
  if (tid == 0){
    float a0 = p1b[0], a1 = p1b[1];
    for (int w = 0; w < 8; ++w){ a0 += red0[w]; a1 += red1[w]; }
    float m = fmaxf(a0, a1), e0 = __expf(a0 - m), e1 = __expf(a1 - m), inv = 1.f/(e0 + e1);
    pr1[b*2] = e0 * inv; pr1[b*2 + 1] = e1 * inv;
  }
}

// ------------------------------------------------------------------ mix -> bf16
__global__ void mix_kernel(const float* __restrict__ gx, const float* __restrict__ lx,
                           const float* __restrict__ pr1, __bf16* __restrict__ mixbf){
  int i = blockIdx.x * blockDim.x + threadIdx.x;
  if (i >= Bb * Ss * Ee) return;
  int row = i >> 8;           // /E
  int b   = row / Ss;
  mixbf[i] = (__bf16)(pr1[b*2] * gx[i] + pr1[b*2 + 1] * lx[i]);
}

// ------------------------------------------------------------------ pr2 reduction
// Reads G/L once; p2_W chunk staged in LDS and shared across all batches.
__global__ __launch_bounds__(256)
void pr2_reduce_kernel(const float* __restrict__ G, const float* __restrict__ L,
                       const float* __restrict__ p2W, float* __restrict__ acc){
  __shared__ float sp[4][2048];
  __shared__ float red0[8], red1[8];
  const int tid = threadIdx.x;
  const int cbase = blockIdx.x * 2048;
  int len = SV - cbase; if (len > 2048) len = 2048;
  for (int i = tid; i < len; i += 256){
    sp[0][i] = p2W[cbase + i];            // row0 vs global logits
    sp[1][i] = p2W[SV + cbase + i];       // row0 vs local logits
    sp[2][i] = p2W[2*SV + cbase + i];     // row1 vs global
    sp[3][i] = p2W[3*SV + cbase + i];     // row1 vs local
  }
  __syncthreads();
  for (int b = 0; b < Bb; ++b){
    const float* Gb = G + (size_t)b * SV + cbase;
    const float* Lb = L + (size_t)b * SV + cbase;
    float d0 = 0.f, d1 = 0.f;
    for (int i = tid; i < len; i += 256){
      float g = Gb[i], l = Lb[i];
      d0 += g * sp[0][i] + l * sp[1][i];
      d1 += g * sp[2][i] + l * sp[3][i];
    }
    for (int o = 16; o > 0; o >>= 1){ d0 += __shfl_xor(d0, o, 32); d1 += __shfl_xor(d1, o, 32); }
    if ((tid & 31) == 0){ red0[tid >> 5] = d0; red1[tid >> 5] = d1; }
    __syncthreads();
    if (tid == 0){
      float a0 = 0.f, a1 = 0.f;
      for (int w = 0; w < 8; ++w){ a0 += red0[w]; a1 += red1[w]; }
      atomicAdd(&acc[b*2], a0);
      atomicAdd(&acc[b*2 + 1], a1);
    }
    __syncthreads();
  }
}

// ------------------------------------------------------------------ final mix + transpose [B,S,V]->[B,V,S]
__global__ __launch_bounds__(256)
void out_mix_kernel(const float* __restrict__ G, const float* __restrict__ L,
                    const float* __restrict__ acc, const float* __restrict__ p2b,
                    float* __restrict__ out){
  __shared__ float pr[2];
  const int b = blockIdx.y;
  if (threadIdx.x == 0){
    float a0 = acc[b*2] + p2b[0];
    float a1 = acc[b*2 + 1] + p2b[1];
    float m = fmaxf(a0, a1), e0 = __expf(a0 - m), e1 = __expf(a1 - m), inv = 1.f/(e0 + e1);
    pr[0] = e0 * inv; pr[1] = e1 * inv;       // pr[0] pairs with GLOBAL logits
  }
  __syncthreads();
  int j = blockIdx.x * 256 + threadIdx.x;     // j = v*S + s  -> coalesced stores
  if (j >= SV) return;
  int v = j / Ss;
  int s = j - v * Ss;
  size_t src = ((size_t)b * Ss + s) * Vv + v;
  out[(size_t)b * SV + j] = pr[0] * G[src] + pr[1] * L[src];
}

// ==================================================================
extern "C" void kernel_launch(void* const* d_in, const int* in_sizes, int n_in,
                              void* d_out, int out_size, void* d_ws, size_t ws_size,
                              hipStream_t stream) {
  (void)in_sizes; (void)n_in; (void)out_size; (void)ws_size;
  const int*   x    = (const int*)  d_in[0];
  const float* emb  = (const float*)d_in[1];
  const float* gWih = (const float*)d_in[2];
  const float* gWhh = (const float*)d_in[3];
  const float* gB   = (const float*)d_in[4];
  const float* lWih = (const float*)d_in[5];
  const float* lWhh = (const float*)d_in[6];
  const float* lB   = (const float*)d_in[7];
  const float* gl1W = (const float*)d_in[8];
  const float* gl1b = (const float*)d_in[9];
  const float* ll1W = (const float*)d_in[10];
  const float* ll1b = (const float*)d_in[11];
  const float* gl2W = (const float*)d_in[12];
  const float* gl2b = (const float*)d_in[13];
  const float* ll2W = (const float*)d_in[14];
  const float* ll2b = (const float*)d_in[15];
  const float* plW  = (const float*)d_in[16];
  const float* plb  = (const float*)d_in[17];
  const float* p1W  = (const float*)d_in[18];
  const float* p1b  = (const float*)d_in[19];
  const float* p2W  = (const float*)d_in[20];
  const float* p2b  = (const float*)d_in[21];
  float* out = (float*)d_out;

  char* ws = (char*)d_ws;
  size_t off = 0;
  auto alloc = [&](size_t bytes) -> void* {
    void* p = ws + off;
    off += (bytes + 255) & ~(size_t)255;
    return p;
  };

  float*  e32   = (float*) alloc((size_t)Bb*Ss*Ee*4);
  __bf16* ebf   = (__bf16*)alloc((size_t)Bb*Ss*Ee*2);
  __bf16* wihg  = (__bf16*)alloc((size_t)H4*Ee*2);
  __bf16* wihl  = (__bf16*)alloc((size_t)H4*Ee*2);
  __bf16* whhg  = (__bf16*)alloc((size_t)H4*Hh*2);
  __bf16* whhl  = (__bf16*)alloc((size_t)H4*Hh*2);
  __bf16* w1g   = (__bf16*)alloc((size_t)Ee*Hh*2);
  __bf16* w1l   = (__bf16*)alloc((size_t)Ee*Hh*2);
  __bf16* w2g   = (__bf16*)alloc((size_t)Vv*Ee*2);
  __bf16* w2l   = (__bf16*)alloc((size_t)Vv*Ee*2);
  float*  ging  = (float*) alloc((size_t)Bb*Ss*H4*4);
  float*  ginl  = (float*) alloc((size_t)Bb*Ss*H4*4);
  float*  gatg  = (float*) alloc((size_t)Bb*H4*4);
  float*  gatl  = (float*) alloc((size_t)Bb*H4*4);
  float*  gc    = (float*) alloc((size_t)Bb*Hh*4);
  float*  lc    = (float*) alloc((size_t)Bb*Hh*4);
  __bf16* ghbf  = (__bf16*)alloc((size_t)Bb*Hh*2);
  __bf16* lhbf  = (__bf16*)alloc((size_t)Bb*Hh*2);
  __bf16* hsbf  = (__bf16*)alloc((size_t)Bb*Ss*Hh*2);
  float*  gx    = (float*) alloc((size_t)Bb*Ss*Ee*4);
  float*  lx    = (float*) alloc((size_t)Bb*Ss*Ee*4);
  __bf16* mixbf = (__bf16*)alloc((size_t)Bb*Ss*Ee*2);
  float*  pr1   = (float*) alloc((size_t)Bb*2*4);
  float*  acc   = (float*) alloc((size_t)Bb*2*4);
  float*  Gl    = (float*) alloc((size_t)Bb*Ss*Vv*4);
  float*  Ll    = (float*) alloc((size_t)Bb*Ss*Vv*4);

  // ---- init + embed + weight casts
  init_zero_kernel<<<(Bb*Hh + 255)/256, 256, 0, stream>>>(gc, lc, ghbf, lhbf, acc);
  embed_kernel<<<Bb*Ss, Ee, 0, stream>>>(x, emb, e32, ebf);
  auto cvt = [&](const float* s, __bf16* d, int n){
    cvt_bf16_kernel<<<(n + 255)/256, 256, 0, stream>>>(s, d, n);
  };
  cvt(gWih, wihg, H4*Ee);  cvt(lWih, wihl, H4*Ee);
  cvt(gWhh, whhg, H4*Hh);  cvt(lWhh, whhl, H4*Hh);
  cvt(gl1W, w1g, Ee*Hh);   cvt(ll1W, w1l, Ee*Hh);
  cvt(gl2W, w2g, Vv*Ee);   cvt(ll2W, w2l, Vv*Ee);

  // ---- input projections for ALL steps (hoisted out of the scan)
  // gin[cell][B*S, 4H] = e @ Wih.T    K=256, 200x128 tiles -> 6400 waves
  gemm_bf16_wmma<Ee, 4, false, false><<<dim3(800, 1, 2), 256, 0, stream>>>(
      ebf, nullptr, Ee, wihg, wihl, Ee,
      nullptr, nullptr, nullptr, nullptr, 0,
      ging, ginl, H4, 200, 0);

  // ---- sequential scan: recurrent GEMM + fused cell update
  for (int t = 0; t < Ss; ++t){
    // gates[cell][B,4H] = h_bf @ Whh.T + gin[cell][b*S+t,:] + b  K=512, 8x128 tiles -> 256 waves
    gemm_bf16_wmma<Hh, 4, true, true><<<dim3(32, 1, 2), 256, 0, stream>>>(
        ghbf, lhbf, Hh, whhg, whhl, Hh,
        gB, lB, ging + (size_t)t*H4, ginl + (size_t)t*H4, Ss*H4,
        gatg, gatl, H4, 8, 0);
    lstm_cell_update<<<Bb, 256, 0, stream>>>(
        gatg, gatl, e32, plW, plb, gc, lc, ghbf, lhbf, hsbf, t);
  }

  // ---- hidden-state projections: [3200,512] x [512,256]  K=512, 200x16 tiles -> 800 waves
  gemm_bf16_wmma<Hh, 4, true, false><<<dim3(100, 1, 2), 256, 0, stream>>>(
      hsbf, nullptr, Hh, w1g, w1l, Hh,
      gl1b, ll1b, nullptr, nullptr, 0,
      gx, lx, Ee, 200, 0);

  pr1_kernel<<<Bb, 256, 0, stream>>>(gx, lx, p1W, p1b, pr1);
  mix_kernel<<<(Bb*Ss*Ee + 255)/256, 256, 0, stream>>>(gx, lx, pr1, mixbf);

  // ---- head GEMMs: [3200,256] x [256,10000]  K=256
  // main: 624 of 625 column tiles (156 groups of 4) -> 200*156 = 31200 waves
  gemm_bf16_wmma<Ee, 4, true, false><<<dim3(3900, 1, 2), 256, 0, stream>>>(
      mixbf, nullptr, Ee, w2g, w2l, Ee,
      gl2b, ll2b, nullptr, nullptr, 0,
      Gl, Ll, Vv, 200, 0);
  // tail: last 16 columns (nbase = 9984) -> 200 waves
  gemm_bf16_wmma<Ee, 1, true, false><<<dim3(25, 1, 2), 256, 0, stream>>>(
      mixbf, nullptr, Ee, w2g, w2l, Ee,
      gl2b, ll2b, nullptr, nullptr, 0,
      Gl, Ll, Vv, 200, 9984);

  // ---- pr2 dots + final mix/transpose
  pr2_reduce_kernel<<<(SV + 2047)/2048, 256, 0, stream>>>(Gl, Ll, p2W, acc);
  out_mix_kernel<<<dim3((SV + 255)/256, Bb), 256, 0, stream>>>(Gl, Ll, acc, p2b, out);
}
